// ChessTransformer_24713241822030
// MI455X (gfx1250) — matrix operand order repaired and checked by reading, not compiled, not run on hardware
//
#include <hip/hip_runtime.h>
#include <hip/hip_bf16.h>
#include <math.h>

typedef __bf16 bf16;
typedef __attribute__((ext_vector_type(16))) __bf16 bf16x16;
typedef __attribute__((ext_vector_type(8)))  __bf16 bf16x8;
typedef __attribute__((ext_vector_type(8)))  float  f32x8;

#define DMODEL 1024
#define NTOK   71
#define NLAYER 12
#define NBATCH 1024

// ---------------------------------------------------------------------------
// Block-wide sum reduction (wave32), sm must hold >= 8 floats of LDS.
// ---------------------------------------------------------------------------
__device__ __forceinline__ float block_sum(float v, float* sm) {
    const int lane = threadIdx.x & 31;
    const int wave = threadIdx.x >> 5;
    #pragma unroll
    for (int off = 16; off; off >>= 1) v += __shfl_down(v, off, 32);
    if (lane == 0) sm[wave] = v;
    __syncthreads();
    float r = (threadIdx.x < 8) ? sm[threadIdx.x] : 0.0f;
    if (wave == 0) {
        #pragma unroll
        for (int off = 4; off; off >>= 1) r += __shfl_down(r, off, 32);
        if (lane == 0) sm[0] = r;
    }
    __syncthreads();
    float out = sm[0];
    __syncthreads();
    return out;
}

// In-place LayerNorm of one 1024-float row, 256 threads.
__device__ __forceinline__ void ln_row_inplace(float* rowp, const float* g,
                                               const float* bta, float* sm) {
    const int tid = threadIdx.x;
    float v[4];
    #pragma unroll
    for (int j = 0; j < 4; ++j) v[j] = rowp[tid + j * 256];
    float mean = block_sum(v[0] + v[1] + v[2] + v[3], sm) * (1.0f / 1024.0f);
    float sq = 0.f;
    #pragma unroll
    for (int j = 0; j < 4; ++j) { float d = v[j] - mean; sq += d * d; }
    float var = block_sum(sq, sm) * (1.0f / 1024.0f);
    float rstd = rsqrtf(var + 1e-5f);
    #pragma unroll
    for (int j = 0; j < 4; ++j) {
        int d = tid + j * 256;
        rowp[d] = (v[j] - mean) * rstd * g[d] + bta[d];
    }
}

// ---------------------------------------------------------------------------
// Weight prep: f32 -> bf16 straight convert (grid-stride).
// ---------------------------------------------------------------------------
__global__ void convert_bf16_kernel(const float* __restrict__ s,
                                    bf16* __restrict__ d, size_t n) {
    size_t i = (size_t)blockIdx.x * blockDim.x + threadIdx.x;
    size_t stride = (size_t)gridDim.x * blockDim.x;
    for (; i < n; i += stride) d[i] = (bf16)s[i];
}

// qkv planes are [K_in, N_out]; transpose to [N_out, K_in] bf16 so the WMMA
// B fragment (one weight row per lane) is contiguous in K.
__global__ void transpose_conv_kernel(const float* __restrict__ src,
                                      bf16* __restrict__ dst) {
    __shared__ float tile[32][33];
    const int plane = blockIdx.z;                       // l*3 + n
    const float* S = src + (size_t)plane * DMODEL * DMODEL;
    bf16* D = dst + (size_t)plane * DMODEL * DMODEL;
    const int i0 = blockIdx.y * 32;                     // input row (d_in)
    const int o0 = blockIdx.x * 32;                     // input col (d_out)
    for (int r = threadIdx.y; r < 32; r += 8)
        tile[r][threadIdx.x] = S[(size_t)(i0 + r) * DMODEL + o0 + threadIdx.x];
    __syncthreads();
    for (int r = threadIdx.y; r < 32; r += 8)
        D[(size_t)(o0 + r) * DMODEL + i0 + threadIdx.x] = (bf16)tile[threadIdx.x][r];
}

// ---------------------------------------------------------------------------
// Embedding + LayerNorm.  grid = (71, B), block = 256.
// ---------------------------------------------------------------------------
__global__ __launch_bounds__(256) void embed_kernel(
    const int* __restrict__ fen, const int* __restrict__ mv,
    const float* __restrict__ rank_emb, const float* __restrict__ file_emb,
    const float* __restrict__ fen_emb, const float* __restrict__ move_emb,
    const float* __restrict__ g, const float* __restrict__ bta,
    float* __restrict__ X, bf16* __restrict__ XB) {
    __shared__ float sm[8];
    const int t = blockIdx.x;
    const int b = blockIdx.y;
    const int tid = threadIdx.x;
    const size_t row = (size_t)b * NTOK + t;
    float v[4];
    #pragma unroll
    for (int j = 0; j < 4; ++j) {
        const int d = tid + j * 256;
        float val;
        if (t < 64) {
            int f1 = fen[b * 133 + t];
            int f2 = fen[b * 133 + 64 + t];
            val = 0.5f * (fen_emb[(size_t)f1 * DMODEL + d] +
                          fen_emb[(size_t)f2 * DMODEL + d] +
                          rank_emb[(size_t)(t >> 3) * DMODEL + d] +
                          file_emb[(size_t)(t & 7) * DMODEL + d]);
        } else if (t < 69) {
            int f = fen[b * 133 + 64 + t];  // tokens 128..132
            val = fen_emb[(size_t)f * DMODEL + d];
        } else {
            int j2 = t - 69;
            int sq = mv[b * 2 + j2];
            val = (rank_emb[(size_t)(sq >> 3) * DMODEL + d] +
                   file_emb[(size_t)(sq & 7) * DMODEL + d] +
                   move_emb[(size_t)j2 * DMODEL + d]) * 0.58f;
        }
        v[j] = val;
    }
    float mean = block_sum(v[0] + v[1] + v[2] + v[3], sm) * (1.0f / 1024.0f);
    float sq = 0.f;
    #pragma unroll
    for (int j = 0; j < 4; ++j) { float d = v[j] - mean; sq += d * d; }
    float var = block_sum(sq, sm) * (1.0f / 1024.0f);
    float rstd = rsqrtf(var + 1e-5f);
    #pragma unroll
    for (int j = 0; j < 4; ++j) {
        int d = tid + j * 256;
        float o = (v[j] - mean) * rstd * g[d] + bta[d];
        X[row * DMODEL + d] = o;
        XB[row * DMODEL + d] = (bf16)o;
    }
}

// ---------------------------------------------------------------------------
// bf16 WMMA GEMM:  C[M,N] = A[M,K] * W[N,K]^T   (both K-contiguous)
// 128x128 tile / block (256 thr = 8 waves, 2x4), K-step 32,
// double-buffered LDS (one barrier / K-step, global prefetch overlaps WMMA).
// act: 0 = none, 1 = leaky_relu(0.2). Optional f32 and/or bf16 outputs.
// ---------------------------------------------------------------------------
__global__ __launch_bounds__(256) void gemm_bf16_kernel(
    const bf16* __restrict__ A, const bf16* __restrict__ W,
    float* __restrict__ Cf, bf16* __restrict__ Cb,
    int M, int N, int K, int act) {
    __shared__ __align__(16) bf16 As[2][128][40];   // 32 K + pad (80B rows)
    __shared__ __align__(16) bf16 Bs[2][128][40];
    const int m0 = blockIdx.y * 128;
    const int n0 = blockIdx.x * 128;
    const int tid  = threadIdx.x;
    const int lane = tid & 31;
    const int wave = tid >> 5;
    const int wm = wave & 1;        // 0..1 -> 64 rows each
    const int wn = wave >> 1;       // 0..3 -> 32 cols each
    const int lr = lane & 15;
    const int lh = lane >> 4;

    // per-thread global tile pointers: 512 (row, 16B-chunk) pairs per array
    const int r0 = tid >> 2,        c0 = tid & 3;
    const int r1 = (tid + 256) >> 2, c1 = (tid + 256) & 3;
    const bf16* gA0 = A + (size_t)(m0 + r0) * K + c0 * 8;
    const bf16* gA1 = A + (size_t)(m0 + r1) * K + c1 * 8;
    const bf16* gB0 = W + (size_t)(n0 + r0) * K + c0 * 8;
    const bf16* gB1 = W + (size_t)(n0 + r1) * K + c1 * 8;

    f32x8 acc[4][2];
    #pragma unroll
    for (int tm = 0; tm < 4; ++tm)
        #pragma unroll
        for (int tn = 0; tn < 2; ++tn)
            #pragma unroll
            for (int r = 0; r < 8; ++r) acc[tm][tn][r] = 0.0f;

    // prologue: fill buffer 0
    {
        uint4 a0 = *(const uint4*)gA0;
        uint4 a1 = *(const uint4*)gA1;
        uint4 b0 = *(const uint4*)gB0;
        uint4 b1 = *(const uint4*)gB1;
        *(uint4*)&As[0][r0][c0 * 8] = a0;
        *(uint4*)&As[0][r1][c1 * 8] = a1;
        *(uint4*)&Bs[0][r0][c0 * 8] = b0;
        *(uint4*)&Bs[0][r1][c1 * 8] = b1;
    }
    __syncthreads();

    const int nsteps = K >> 5;
    for (int s = 0; s < nsteps; ++s) {
        const int cur = s & 1;
        const bool more = (s + 1) < nsteps;

        // issue next-step global loads first (latency hidden under WMMA)
        uint4 pa0, pa1, pb0, pb1;
        if (more) {
            const int k1 = (s + 1) << 5;
            pa0 = *(const uint4*)(gA0 + k1);
            pa1 = *(const uint4*)(gA1 + k1);
            pb0 = *(const uint4*)(gB0 + k1);
            pb1 = *(const uint4*)(gB1 + k1);
        }

        // fragment loads (batched so ds waits drain under WMMA issue)
        bf16x16 bfrag[2], afrag[4];
        #pragma unroll
        for (int tn = 0; tn < 2; ++tn) {
            const int n = wn * 32 + tn * 16 + lr;
            bf16x8 b0 = *(const bf16x8*)&Bs[cur][n][lh * 8];
            bf16x8 b1 = *(const bf16x8*)&Bs[cur][n][16 + lh * 8];
            bfrag[tn] = __builtin_shufflevector(b0, b1, 0, 1, 2, 3, 4, 5, 6, 7,
                                                8, 9, 10, 11, 12, 13, 14, 15);
        }
        #pragma unroll
        for (int tm = 0; tm < 4; ++tm) {
            const int row = wm * 64 + tm * 16 + lr;
            bf16x8 a0 = *(const bf16x8*)&As[cur][row][lh * 8];       // K 8h..
            bf16x8 a1 = *(const bf16x8*)&As[cur][row][16 + lh * 8];  // K 16+8h..
            afrag[tm] = __builtin_shufflevector(a0, a1, 0, 1, 2, 3, 4, 5, 6, 7,
                                                8, 9, 10, 11, 12, 13, 14, 15);
        }
        #pragma unroll
        for (int tm = 0; tm < 4; ++tm)
            #pragma unroll
            for (int tn = 0; tn < 2; ++tn)
                acc[tm][tn] = __builtin_amdgcn_wmma_f32_16x16x32_bf16(
                    false, afrag[tm], false, bfrag[tn], (short)0, acc[tm][tn],
                    false, false);

        // stage next tile into the other buffer; single barrier per step
        if (more) {
            const int nxt = cur ^ 1;
            *(uint4*)&As[nxt][r0][c0 * 8] = pa0;
            *(uint4*)&As[nxt][r1][c1 * 8] = pa1;
            *(uint4*)&Bs[nxt][r0][c0 * 8] = pb0;
            *(uint4*)&Bs[nxt][r1][c1 * 8] = pb1;
            __syncthreads();
        }
    }

    // epilogue
    #pragma unroll
    for (int tm = 0; tm < 4; ++tm) {
        #pragma unroll
        for (int tn = 0; tn < 2; ++tn) {
            const int n = n0 + wn * 32 + tn * 16 + lr;
            const int mb = m0 + wm * 64 + tm * 16 + lh * 8;
            #pragma unroll
            for (int r = 0; r < 8; ++r) {
                float v = acc[tm][tn][r];
                if (act) v = (v > 0.0f) ? v : 0.2f * v;
                const size_t idx = (size_t)(mb + r) * N + n;
                if (Cf) Cf[idx] = v;
                if (Cb) Cb[idx] = (bf16)v;
            }
        }
    }
}

// ---------------------------------------------------------------------------
// LayerNorm q and k rows of Y [M,3072] in place.  grid = (71, B).
// ---------------------------------------------------------------------------
__global__ __launch_bounds__(256) void qk_ln_kernel(
    float* __restrict__ Y, const float* __restrict__ gq,
    const float* __restrict__ bq, const float* __restrict__ gk,
    const float* __restrict__ bk) {
    __shared__ float sm[8];
    const size_t row = (size_t)blockIdx.y * NTOK + blockIdx.x;
    float* base = Y + row * 3072;
    ln_row_inplace(base, gq, bq, sm);
    ln_row_inplace(base + DMODEL, gk, bk, sm);
}

// ---------------------------------------------------------------------------
// Attention: grid = (ceil(71/8)=9, B), block = 256.
// Y rows: [q | k | v] each 1024.  O = softmax(qk^T/32) v,  [M,1024] f32.
// ---------------------------------------------------------------------------
__global__ __launch_bounds__(256) void attn_kernel(const float* __restrict__ Y,
                                                   float* __restrict__ O) {
    __shared__ float qs[8][DMODEL];
    __shared__ float ss[8][72];
    const int t0 = blockIdx.x * 8;
    const int b = blockIdx.y;
    const int tid = threadIdx.x;
    const int lane = tid & 31;
    const int wave = tid >> 5;
    const size_t rowbase = (size_t)b * NTOK * 3072;

    for (int i = tid; i < 8 * DMODEL; i += 256) {
        int qi = i >> 10, d = i & 1023;
        int t = t0 + qi;
        qs[qi][d] = (t < NTOK) ? Y[rowbase + (size_t)t * 3072 + d] : 0.0f;
    }
    __syncthreads();

    // scores: wave `w` owns query row qi = w
    const int qi = wave;
    for (int s = 0; s < NTOK; ++s) {
        const float* krow = Y + rowbase + (size_t)s * 3072 + DMODEL;
        float p = 0.0f;
        #pragma unroll
        for (int j = 0; j < 32; ++j) {
            int d = lane + j * 32;
            p += qs[qi][d] * krow[d];
        }
        #pragma unroll
        for (int off = 16; off; off >>= 1) p += __shfl_down(p, off, 32);
        if (lane == 0) ss[qi][s] = p * 0.03125f;   // 1/sqrt(1024)
    }
    // wave-local softmax over 71 scores
    float v0 = ss[qi][lane];
    float v1 = (lane + 32 < NTOK) ? ss[qi][lane + 32] : -3.4e38f;
    float v2 = (lane + 64 < NTOK) ? ss[qi][lane + 64] : -3.4e38f;
    float mx = fmaxf(v0, fmaxf(v1, v2));
    #pragma unroll
    for (int off = 16; off; off >>= 1) mx = fmaxf(mx, __shfl_xor(mx, off, 32));
    float e0 = expf(v0 - mx);
    float e1 = (lane + 32 < NTOK) ? expf(v1 - mx) : 0.0f;
    float e2 = (lane + 64 < NTOK) ? expf(v2 - mx) : 0.0f;
    float sum = e0 + e1 + e2;
    #pragma unroll
    for (int off = 16; off; off >>= 1) sum += __shfl_xor(sum, off, 32);
    float inv = 1.0f / sum;
    ss[qi][lane] = e0 * inv;
    if (lane + 32 < NTOK) ss[qi][lane + 32] = e1 * inv;
    if (lane + 64 < NTOK) ss[qi][lane + 64] = e2 * inv;
    __syncthreads();

    // out = att @ v : thread owns 4 consecutive d's for all 8 queries
    float acc[8][4];
    #pragma unroll
    for (int q2 = 0; q2 < 8; ++q2)
        #pragma unroll
        for (int j = 0; j < 4; ++j) acc[q2][j] = 0.0f;
    const int d0 = tid * 4;
    for (int s = 0; s < NTOK; ++s) {
        const float4 vv =
            *(const float4*)(Y + rowbase + (size_t)s * 3072 + 2048 + d0);
        #pragma unroll
        for (int q2 = 0; q2 < 8; ++q2) {
            float a = ss[q2][s];
            acc[q2][0] += a * vv.x; acc[q2][1] += a * vv.y;
            acc[q2][2] += a * vv.z; acc[q2][3] += a * vv.w;
        }
    }
    #pragma unroll
    for (int q2 = 0; q2 < 8; ++q2) {
        int t = t0 + q2;
        if (t < NTOK) {
            float4 o = make_float4(acc[q2][0], acc[q2][1], acc[q2][2], acc[q2][3]);
            *(float4*)(O + ((size_t)b * NTOK + t) * DMODEL + d0) = o;
        }
    }
}

// ---------------------------------------------------------------------------
// x = 0.7*(x + LN(src));  writes f32 X and bf16 XB.  grid = (71, B).
// ---------------------------------------------------------------------------
__global__ __launch_bounds__(256) void ln_res_kernel(
    const float* __restrict__ S, const float* __restrict__ g,
    const float* __restrict__ bta, float* __restrict__ X,
    bf16* __restrict__ XB) {
    __shared__ float sm[8];
    const size_t row = (size_t)blockIdx.y * NTOK + blockIdx.x;
    const int tid = threadIdx.x;
    float v[4];
    #pragma unroll
    for (int j = 0; j < 4; ++j) v[j] = S[row * DMODEL + tid + j * 256];
    float mean = block_sum(v[0] + v[1] + v[2] + v[3], sm) * (1.0f / 1024.0f);
    float sq = 0.f;
    #pragma unroll
    for (int j = 0; j < 4; ++j) { float d = v[j] - mean; sq += d * d; }
    float var = block_sum(sq, sm) * (1.0f / 1024.0f);
    float rstd = rsqrtf(var + 1e-5f);
    #pragma unroll
    for (int j = 0; j < 4; ++j) {
        int d = tid + j * 256;
        float h = (v[j] - mean) * rstd * g[d] + bta[d];
        float x = 0.7f * (X[row * DMODEL + d] + h);
        X[row * DMODEL + d] = x;
        XB[row * DMODEL + d] = (bf16)x;
    }
}

// ---------------------------------------------------------------------------
// Head: out[b] = sigmoid(x[b,69:71,:] . w + bias).  grid = B, block = 256.
// (rows 69 and 70 are contiguous -> a single 2048-length dot)
// ---------------------------------------------------------------------------
__global__ __launch_bounds__(256) void head_kernel(const float* __restrict__ X,
                                                   const float* __restrict__ w,
                                                   const float* __restrict__ bias,
                                                   float* __restrict__ out) {
    __shared__ float sm[8];
    const int b = blockIdx.x;
    const float* x0 = X + ((size_t)b * NTOK + 69) * DMODEL;  // spans rows 69,70
    float p = 0.0f;
    #pragma unroll
    for (int i = 0; i < 8; ++i) {
        int j = threadIdx.x + i * 256;
        p += x0[j] * w[j];
    }
    float s = block_sum(p, sm);
    if (threadIdx.x == 0) out[b] = 1.0f / (1.0f + expf(-(s + bias[0])));
}

// ---------------------------------------------------------------------------
extern "C" void kernel_launch(void* const* d_in, const int* in_sizes, int n_in,
                              void* d_out, int out_size, void* d_ws, size_t ws_size,
                              hipStream_t stream) {
    (void)in_sizes; (void)n_in; (void)out_size; (void)ws_size;
    const int M = NBATCH * NTOK;  // 72704 = 568 * 128

    const int*   fen      = (const int*)d_in[0];
    const int*   mv       = (const int*)d_in[1];
    const float* rank_emb = (const float*)d_in[2];
    const float* file_emb = (const float*)d_in[3];
    const float* fen_emb  = (const float*)d_in[4];
    const float* move_emb = (const float*)d_in[5];
    const float* ln_emb_g = (const float*)d_in[6];
    const float* ln_emb_b = (const float*)d_in[7];
    const float* qkv      = (const float*)d_in[8];
    const float* ln_q_g   = (const float*)d_in[9];
    const float* ln_q_b   = (const float*)d_in[10];
    const float* ln_k_g   = (const float*)d_in[11];
    const float* ln_k_b   = (const float*)d_in[12];
    const float* ln_attn_g= (const float*)d_in[13];
    const float* ln_attn_b= (const float*)d_in[14];
    const float* lin1_w   = (const float*)d_in[15];
    const float* lin2_w   = (const float*)d_in[16];
    const float* ln_ff_g  = (const float*)d_in[17];
    const float* ln_ff_b  = (const float*)d_in[18];
    const float* out_w    = (const float*)d_in[19];
    const float* out_b    = (const float*)d_in[20];

    // workspace carve (256B aligned)
    char* wp = (char*)d_ws;
    auto carve = [&](size_t bytes) -> void* {
        char* r = wp;
        wp += (bytes + 255) & ~(size_t)255;
        return (void*)r;
    };
    bf16*  WQKV = (bf16*)carve((size_t)NLAYER * 3 * DMODEL * DMODEL * sizeof(bf16));
    bf16*  W1   = (bf16*)carve((size_t)NLAYER * 2048 * 1024 * sizeof(bf16));
    bf16*  W2   = (bf16*)carve((size_t)NLAYER * 1024 * 2048 * sizeof(bf16));
    float* X    = (float*)carve((size_t)M * DMODEL * sizeof(float));
    bf16*  XB   = (bf16*)carve((size_t)M * DMODEL * sizeof(bf16));
    float* Y    = (float*)carve((size_t)M * 3072 * sizeof(float));
    bf16*  H    = (bf16*)carve((size_t)M * 2048 * sizeof(bf16));
    float* SCR  = (float*)carve((size_t)M * DMODEL * sizeof(float));

    // weight prep (every call; deterministic)
    convert_bf16_kernel<<<8192, 256, 0, stream>>>(lin1_w, W1,
        (size_t)NLAYER * 2048 * 1024);
    convert_bf16_kernel<<<8192, 256, 0, stream>>>(lin2_w, W2,
        (size_t)NLAYER * 1024 * 2048);
    transpose_conv_kernel<<<dim3(32, 32, NLAYER * 3), dim3(32, 8), 0, stream>>>(
        qkv, WQKV);

    // embedding + LN
    embed_kernel<<<dim3(NTOK, NBATCH), 256, 0, stream>>>(
        fen, mv, rank_emb, file_emb, fen_emb, move_emb, ln_emb_g, ln_emb_b,
        X, XB);

    for (int l = 0; l < NLAYER; ++l) {
        const bf16* wq = WQKV + (size_t)l * 3 * DMODEL * DMODEL;
        const bf16* w1 = W1 + (size_t)l * 2048 * 1024;
        const bf16* w2 = W2 + (size_t)l * 1024 * 2048;

        // QKV: [M,1024] x [3072,1024]^T -> Y [M,3072]
        gemm_bf16_kernel<<<dim3(3072 / 128, M / 128), 256, 0, stream>>>(
            XB, wq, Y, nullptr, M, 3072, 1024, 0);

        qk_ln_kernel<<<dim3(NTOK, NBATCH), 256, 0, stream>>>(
            Y, ln_q_g + l * DMODEL, ln_q_b + l * DMODEL,
            ln_k_g + l * DMODEL, ln_k_b + l * DMODEL);

        attn_kernel<<<dim3(9, NBATCH), 256, 0, stream>>>(Y, SCR);

        ln_res_kernel<<<dim3(NTOK, NBATCH), 256, 0, stream>>>(
            SCR, ln_attn_g + l * DMODEL, ln_attn_b + l * DMODEL, X, XB);

        // FFN1: leaky -> bf16 H
        gemm_bf16_kernel<<<dim3(2048 / 128, M / 128), 256, 0, stream>>>(
            XB, w1, nullptr, H, M, 2048, 1024, 1);
        // FFN2: leaky -> f32 SCR
        gemm_bf16_kernel<<<dim3(1024 / 128, M / 128), 256, 0, stream>>>(
            H, w2, SCR, nullptr, M, 1024, 2048, 1);

        ln_res_kernel<<<dim3(NTOK, NBATCH), 256, 0, stream>>>(
            SCR, ln_ff_g + l * DMODEL, ln_ff_b + l * DMODEL, X, XB);
    }

    head_kernel<<<NBATCH, 256, 0, stream>>>(X, out_w, out_b, (float*)d_out);
}
